// OptimizedHierarchicalEncoder_23613730193796
// MI455X (gfx1250) — matrix operand order repaired and checked by reading, not compiled, not run on hardware
//
#include <hip/hip_runtime.h>
#include <hip/hip_bf16.h>

typedef __attribute__((ext_vector_type(16))) _Float16 v16h;
typedef __attribute__((ext_vector_type(8)))  _Float16 v8h;
typedef __attribute__((ext_vector_type(8)))  float    v8f;

#define NJ   17
#define HID  64
#define OUTF 128
#define FDIM 384   // 3 levels * (64 Hsum + 64 Z)

// ---- topology constants (precomputed from SUBSETS) ----
__device__ __constant__ int   c_G[3]    = {9, 13, 8};
__device__ __constant__ int   c_S[3]    = {5, 4, 4};
__device__ __constant__ float c_invS[3] = {0.2f, 0.25f, 0.25f};
__device__ __constant__ int c_allnb[3][13] = {
  {0,5,6,7,8,11,12,13,14, 0,0,0,0},
  {0,5,6,7,8,9,10,11,12,13,14,15,16},
  {7,8,9,10,13,14,15,16, 0,0,0,0,0}};
// sub-index of allnb slot, -1 if slot not in sub
__device__ __constant__ int c_subpos[3][13] = {
  { 0, 1, 2,-1,-1, 3, 4,-1,-1, -1,-1,-1,-1},
  {-1,-1,-1, 0, 1,-1,-1,-1,-1,  2, 3,-1,-1},
  {-1,-1, 0, 1,-1,-1, 2, 3, -1,-1,-1,-1,-1}};
__device__ __constant__ int c_subj[3][5]  = {{0,5,6,11,12},{7,8,13,14,0},{9,10,15,16,0}};
__device__ __constant__ int c_subvs[3][5] = {{0,1,2,5,6},  {3,4,9,10,0}, {2,3,6,7,0}};

union F16x16 { v16h v; _Float16 h[16]; };
union F16x8  { v8h  v; _Float16 h[8];  };
union F32x8  { v8f  v; float    f[8];  };

__device__ __forceinline__ v8f wmma_f16(v16h a, v16h b, v8f c) {
  // D = A(16x32 f16) x B(32x16 f16) + C(16x16 f32)
  return __builtin_amdgcn_wmma_f32_16x16x32_f16(
      /*neg_a=*/false, a, /*neg_b=*/false, b,
      /*c_mod=*/(short)0, c, /*reuse_a=*/false, /*reuse_b=*/false);
}

// B fragment from f16 We staged in LDS: rows of the 32-row K step in table
// order for this lane's K-half (kh*8+h, 16+kh*8+h-8), column = this lane's n.
__device__ __forceinline__ v16h load_we_frag(const _Float16* sWe, int krow0, int col) {
  F16x16 B;
#pragma unroll
  for (int h = 0; h < 16; ++h) B.h[h] = sWe[((krow0 & ~31) + ((krow0 >> 4) & 1) * 8 + (h < 8 ? h : 8 + h)) * HID + col];
  return B.v;
}

// Per-joint hidden features m = relu(P@W + b) * s, produced directly in the
// 16-bit A-fragment register layout (lane owns sample r = lane&15; khalf
// selects the K-feature interleave).  A0: features 0..31, A1: 32..63.
__device__ __forceinline__ void compute_m(
    int g, int sG, float mnx, float ivx, float mny, float ivy,
    const float* __restrict__ kp, const float* __restrict__ sc,
    const float* sWl, const int* fofs,
    v16h& A0, v16h& A1, float* mv /*nullable: 32 raw values for Hsum*/)
{
  const float x = kp[sG * NJ * 2 + g * 2 + 0];
  const float y = kp[sG * NJ * 2 + g * 2 + 1];
  const float s = sc[sG * NJ + g];
  const float kx = (x - mnx) * ivx;
  const float ky = (y - mny) * ivy;
  F16x16 a0, a1;
#pragma unroll
  for (int h = 0; h < 16; ++h) {
    const int f  = fofs[h];
    const int f2 = f + 32;
    float m0 = fmaxf(kx * sWl[f]  + ky * sWl[64 + f]  + s * sWl[128 + f]  + sWl[192 + f],  0.f) * s;
    float m1 = fmaxf(kx * sWl[f2] + ky * sWl[64 + f2] + s * sWl[128 + f2] + sWl[192 + f2], 0.f) * s;
    a0.h[h] = (_Float16)m0;
    a1.h[h] = (_Float16)m1;
    if (mv) { mv[h] = m0; mv[16 + h] = m1; }
  }
  A0 = a0.v; A1 = a1.v;
}

__global__ __launch_bounds__(32)
void hier_encoder_wmma(const float* __restrict__ kp, const float* __restrict__ sc,
                       const float* __restrict__ W0, const float* __restrict__ b0,
                       const float* __restrict__ W1, const float* __restrict__ b1,
                       const float* __restrict__ W2, const float* __restrict__ b2,
                       const float* __restrict__ We, const float* __restrict__ be,
                       const float* __restrict__ Wp, const float* __restrict__ bp,
                       float* __restrict__ out)
{
  __shared__ _Float16 sWe[128 * HID];        // 16 KB  edge MLP weights, f16
  // v_g tiles, lane-contiguous: [slot][lane(kh,n)][t][vv] -> 4 x b128 per lane
  __shared__ _Float16 sV[13 * 32 * 32];      // 26 KB
  __shared__ _Float16 sF[16 * FDIM];         // 12 KB  pooled features (A-source)
  __shared__ float    sWl[256];              //  1 KB  level MLP weights + bias

  const int tid = threadIdx.x;
  const int kh  = tid >> 4;      // K-half / row-group selector
  const int n   = tid & 15;      // B/C column, also sample row for A fragments
  const int r   = n;             // local sample owned by this lane (A layout)
  const int sG  = blockIdx.x * 16 + n;   // global sample index
  const int laneV = (kh * 16 + n) * 32;  // this lane's 32-half slice inside a sV slot

  // 16-bit A-fragment feature interleave for this lane
  int fofs[16];
#pragma unroll
  for (int h = 0; h < 16; ++h)
    fofs[h] = (h < 8) ? (kh * 8 + h) : (16 + kh * 8 + (h - 8));

  __builtin_prefetch(Wp + tid * 64, 0, 1);   // global_prefetch_b8: warm Wp into L2

  // stage We -> LDS f16 (coalesced)
  for (int i = tid; i < 128 * HID; i += 32) sWe[i] = (_Float16)We[i];

  // per-sample bbox normalization (min/max over 17 joints, both coords)
  float mnx = 1e30f, mxx = -1e30f, mny = 1e30f, mxy = -1e30f;
#pragma unroll
  for (int j = 0; j < NJ; ++j) {
    const float x = kp[sG * NJ * 2 + j * 2 + 0];
    const float y = kp[sG * NJ * 2 + j * 2 + 1];
    mnx = fminf(mnx, x); mxx = fmaxf(mxx, x);
    mny = fminf(mny, y); mxy = fmaxf(mxy, y);
  }
  const float ivx = 1.f / (mxx - mnx + 1e-6f);
  const float ivy = 1.f / (mxy - mny + 1e-6f);

  float beF[4];
#pragma unroll
  for (int t = 0; t < 4; ++t) beF[t] = be[t * 16 + n];

  __syncthreads();

  for (int lvl = 0; lvl < 3; ++lvl) {
    const int G = c_G[lvl];
    const int S = c_S[lvl];
    const float* Wsel = (lvl == 0) ? W0 : (lvl == 1) ? W1 : W2;
    const float* bsel = (lvl == 0) ? b0 : (lvl == 1) ? b1 : b2;
    for (int i = tid; i < 192; i += 32) sWl[i] = Wsel[i];
    for (int i = tid; i < 64;  i += 32) sWl[192 + i] = bsel[i];
    __syncthreads();

    // ---- phase 1: per joint g in allnb: m_g, Hsum accum, v_g = m_g @ We_bot
    {
      float hs[32];
#pragma unroll
      for (int e = 0; e < 32; ++e) hs[e] = 0.f;

      for (int gi = 0; gi < G; ++gi) {
        const int g  = c_allnb[lvl][gi];
        const int sp = c_subpos[lvl][gi];
        v16h A0, A1; float mv[32];
        compute_m(g, sG, mnx, ivx, mny, ivy, kp, sc, sWl, fofs, A0, A1, mv);
        if (sp >= 0) {
#pragma unroll
          for (int e = 0; e < 32; ++e) hs[e] += mv[e];
        }
#pragma unroll
        for (int t = 0; t < 4; ++t) {
          F32x8 c; c.v = v8f{};
          c.v = wmma_f16(A0, load_we_frag(sWe, 64 +  0 + kh * 16, t * 16 + n), c.v);
          c.v = wmma_f16(A1, load_we_frag(sWe, 64 + 32 + kh * 16, t * 16 + n), c.v);
          F16x8 p;
#pragma unroll
          for (int vv = 0; vv < 8; ++vv) p.h[vv] = (_Float16)c.f[vv];
          *(v8h*)&sV[gi * 1024 + laneV + t * 8] = p.v;   // one ds_store_b128
        }
      }

      // flush Hsum now so hs is dead during phase 2 (VGPR pressure)
#pragma unroll
      for (int h = 0; h < 16; ++h) {
        sF[r * FDIM + lvl * 128 + fofs[h]]      = (_Float16)hs[h];
        sF[r * FDIM + lvl * 128 + fofs[h] + 32] = (_Float16)hs[16 + h];
      }
    }
    __syncthreads();

    // ---- phase 2: per sub joint j: base = u_j - v_j + be, max over k, mean over j
    float Zs[32];
#pragma unroll
    for (int e = 0; e < 32; ++e) Zs[e] = 0.f;

    for (int si = 0; si < S; ++si) {
      const int j  = c_subj[lvl][si];
      const int js = c_subvs[lvl][si];   // j's slot inside allnb
      v16h A0, A1;
      compute_m(j, sG, mnx, ivx, mny, ivy, kp, sc, sWl, fofs, A0, A1, nullptr);

      float base[32];                    // u_j - v_j + be  (single live array)
#pragma unroll
      for (int t = 0; t < 4; ++t) {
        F32x8 c; c.v = v8f{};
        c.v = wmma_f16(A0, load_we_frag(sWe,  0 + kh * 16, t * 16 + n), c.v);
        c.v = wmma_f16(A1, load_we_frag(sWe, 32 + kh * 16, t * 16 + n), c.v);
        F16x8 vj8; vj8.v = *(const v8h*)&sV[js * 1024 + laneV + t * 8];
#pragma unroll
        for (int vv = 0; vv < 8; ++vv)
          base[t * 8 + vv] = c.f[vv] - (float)vj8.h[vv] + beF[t];
      }

      float mxv[32];
#pragma unroll
      for (int e = 0; e < 32; ++e) mxv[e] = 0.f;   // relu(.) >= 0: 0-init exact
      for (int k = 0; k < G; ++k) {
        if (k == js) continue;
        const int kb = k * 1024 + laneV;
#pragma unroll
        for (int t = 0; t < 4; ++t) {
          F16x8 vk8; vk8.v = *(const v8h*)&sV[kb + t * 8];   // one ds_load_b128
#pragma unroll
          for (int vv = 0; vv < 8; ++vv) {
            const float val = fmaxf(base[t * 8 + vv] + (float)vk8.h[vv], 0.f);
            mxv[t * 8 + vv] = fmaxf(mxv[t * 8 + vv], val);
          }
        }
      }
#pragma unroll
      for (int e = 0; e < 32; ++e) Zs[e] += mxv[e];
    }

    // ---- write pooled Z into F (Hsum already flushed after phase 1)
    const float is = c_invS[lvl];
#pragma unroll
    for (int e = 0; e < 32; ++e) {
      const int t = e >> 3, vv = e & 7;
      sF[(vv + 8 * kh) * FDIM + lvl * 128 + 64 + t * 16 + n] = (_Float16)(Zs[e] * is);
    }
    __syncthreads();
  }

  // ---- phase 3: final projection  F(16x384) @ Wp(384x128) + bp
  v8f Co[8] = {};
  for (int kf = 0; kf < 12; ++kf) {
    const int kbase = kf * 32;
    F16x16 Af;
#pragma unroll
    for (int h = 0; h < 8; ++h) Af.h[h]     = sF[r * FDIM + kbase +      kh * 8 + h];
#pragma unroll
    for (int h = 0; h < 8; ++h) Af.h[8 + h] = sF[r * FDIM + kbase + 16 + kh * 8 + h];
#pragma unroll
    for (int nt = 0; nt < 8; ++nt) {
      F16x16 Bw;
#pragma unroll
      for (int h = 0; h < 16; ++h)
        Bw.h[h] = (_Float16)Wp[(kbase + kh * 8 + (h < 8 ? h : 8 + h)) * OUTF + nt * 16 + n];
      Co[nt] = wmma_f16(Af.v, Bw.v, Co[nt]);
    }
  }
#pragma unroll
  for (int nt = 0; nt < 8; ++nt) {
    const float bb = bp[nt * 16 + n];
    F32x8 c; c.v = Co[nt];
#pragma unroll
    for (int vv = 0; vv < 8; ++vv)
      out[(size_t)(blockIdx.x * 16 + vv + 8 * kh) * OUTF + nt * 16 + n] = c.f[vv] + bb;
  }
}

extern "C" void kernel_launch(void* const* d_in, const int* in_sizes, int n_in,
                              void* d_out, int out_size, void* d_ws, size_t ws_size,
                              hipStream_t stream) {
  (void)n_in; (void)out_size; (void)d_ws; (void)ws_size;
  const float* kp = (const float*)d_in[0];
  const float* sc = (const float*)d_in[1];
  const float* W0 = (const float*)d_in[2];
  const float* b0 = (const float*)d_in[3];
  const float* W1 = (const float*)d_in[4];
  const float* b1 = (const float*)d_in[5];
  const float* W2 = (const float*)d_in[6];
  const float* b2 = (const float*)d_in[7];
  const float* We = (const float*)d_in[8];
  const float* be = (const float*)d_in[9];
  const float* Wp = (const float*)d_in[10];
  const float* bp = (const float*)d_in[11];
  float* out = (float*)d_out;

  const int N = in_sizes[0] / (NJ * 2);     // 16384
  const int blocks = N / 16;                // one 16-sample WMMA tile per wave
  hier_encoder_wmma<<<dim3(blocks), dim3(32), 0, stream>>>(
      kp, sc, W0, b0, W1, b1, W2, b2, We, be, Wp, bp, out);
}
